// MultiHeadAttentionPooling1D_17721035064003
// MI455X (gfx1250) — matrix operand order repaired and checked by reading, not compiled, not run on hardware
//
#include <hip/hip_runtime.h>
#include <hip/hip_bf16.h>
#include <math.h>

// Problem constants (match reference)
#define BB 16
#define SS 8192
#define DD 256      // feature dim == HDIMS
#define HH 8        // heads
#define NEG_BIG 1.0e12f

#define MROWS 32            // sequence rows per workgroup (two 16-row M tiles)
#define XBF_STRIDE 264      // bf16 row stride (256 + 8 pad) -> conflict-free b128 loads

typedef __attribute__((ext_vector_type(16))) __bf16        v16bf;
typedef __attribute__((ext_vector_type(8)))  __bf16        v8bf;
typedef __attribute__((ext_vector_type(8)))  float         v8f;
typedef __attribute__((ext_vector_type(4)))  unsigned int  v4u;
typedef __attribute__((ext_vector_type(8)))  int           v8i;
typedef __attribute__((ext_vector_type(4)))  int           v4i;

#if __has_builtin(__builtin_amdgcn_tensor_load_to_lds) && \
    __has_builtin(__builtin_amdgcn_s_wait_tensorcnt)
#define USE_TDM 1
#else
#define USE_TDM 0
#endif

// Branch-free tanh: native V_TANH_F32 on gfx1250 if the builtin exists,
// else exp/rcp based (no saveexec divergence like libm tanhf).
__device__ __forceinline__ float fast_tanh(float x) {
#if __has_builtin(__builtin_amdgcn_tanhf)
    return __builtin_amdgcn_tanhf(x);
#else
    float t = __expf(-2.0f * __builtin_fabsf(x));
    float r = (1.0f - t) * __builtin_amdgcn_rcpf(1.0f + t);
    return __builtin_copysignf(r, x);
#endif
}

// ---------------------------------------------------------------------------
// Kernel 0: k_kernel (D x HDIMS f32, row-major) -> kT (HDIMS x D bf16)
// so a B-matrix fragment is two contiguous 8-element bf16 runs per lane.
// ---------------------------------------------------------------------------
__global__ __launch_bounds__(256) void cvt_kT(const float* __restrict__ kk,
                                              __bf16* __restrict__ kT) {
    int n = blockIdx.x;      // output column of k_kernel (0..255)
    int k = threadIdx.x;     // input  row    of k_kernel (0..255)
    kT[n * DD + k] = (__bf16)kk[k * DD + n];
}

// ---------------------------------------------------------------------------
// Kernel 1: raw scores  w[b][h][s] = tanh(x[b,s,:] @ k_kernel) @ o_kernel
// One workgroup = 32 sequence rows of one batch, full N=256 via WMMA bf16.
// x tile DMA'd to LDS by the TDM, converted ONCE to a bf16 tile in the
// A-fragment layout (two contiguous 8-element runs per lane per k-step),
// then 8 waves x (2 M-frags x 2 N-tiles) x 8 k-steps of WMMA.
// ---------------------------------------------------------------------------
__global__ __launch_bounds__(256) void score_kernel(
    const float*  __restrict__ x,
    const __bf16* __restrict__ kT,
    const float*  __restrict__ o,      // (256 x 8) row-major
    float*        __restrict__ wraw) { // (B x H x S)

    // xs (f32 x-tile) is dead after conversion to xbf; reuse it as hs.
    __shared__ float  xs[MROWS * 257];          // 32 KB: x tile, then tanh(h)
    __shared__ __bf16 xbf[MROWS * XBF_STRIDE];  // 16.5 KB bf16 tile
    __shared__ float  os[DD * HH];              // 8 KB o_kernel
    float* hs = xs;

    const int tid  = threadIdx.x;
    const int tile = blockIdx.x;               // 0..4095
    const int b    = tile >> 8;                // 256 tiles per batch
    const int srow = (tile & 255) << 5;        // first seq row of tile

    const float* xbase = x + (size_t)(b * SS + srow) * DD;

#if USE_TDM
    // TDM: DMA the 32x256 f32 tile into LDS with a 1-DWORD pad every
    // 256 DWORDs (-> 257-float LDS row stride). Only wave 0 issues it.
    if (tid < 32) {
        unsigned long long ga = (unsigned long long)(uintptr_t)xbase;
        v4u g0;
        g0[0] = 1u;                                   // count=1, user D#
        g0[1] = (unsigned)(uintptr_t)xs;              // lds_addr (LDS offset)
        g0[2] = (unsigned)ga;                         // global_addr[31:0]
        g0[3] = (unsigned)((ga >> 32) & 0x1FFFFFFu)   // global_addr[56:32]
              | (2u << 30);                           // type=2 ("image")
        v8i g1;
        g1[0] = (int)((2u << 16)      // data_size = 4 bytes
                    | (1u << 20)      // pad_enable
                    | (7u << 22));    // pad_interval = 256 DWORDs (pad_amount=0 -> 1 DWORD)
        g1[1] = (int)(256u << 16);    // tensor_dim0 = 256 (at bit 48)
        g1[2] = (int)(32u  << 16);    // tensor_dim0 hi = 0 ; tensor_dim1 = 32 (at bit 80)
        g1[3] = (int)(256u << 16);    // tensor_dim1 hi = 0 ; tile_dim0 = 256 (at bit 112)
        g1[4] = (int)MROWS;           // tile_dim1 = 32 ; tile_dim2 = 0
        g1[5] = (int)256;             // tensor_dim0_stride[31:0] = 256
        g1[6] = 0;                    // stride hi / tensor_dim1_stride lo
        g1[7] = 0;
        v4i gz4 = {0, 0, 0, 0};             // groups 2/3 unused (2D tensor)
        v8i gz8 = {0, 0, 0, 0, 0, 0, 0, 0}; // extra group (clang-23 6-arg form)
        __builtin_amdgcn_tensor_load_to_lds(g0, g1, gz4, gz4, gz8, 0);
    }
    __builtin_amdgcn_s_wait_tensorcnt(0);
#else
    for (int i = tid; i < MROWS * DD; i += 256) {
        int r = i >> 8, c = i & 255;
        xs[r * 257 + c] = xbase[r * DD + c];
    }
#endif
    for (int i = tid; i < DD * HH; i += 256) os[i] = o[i];
    __syncthreads();   // x tile visible to all waves

    // One cooperative f32 -> bf16 conversion (replaces 8x redundant
    // per-wave ds_load+cvt work in the WMMA loop).
    for (int i = tid; i < MROWS * DD; i += 256) {
        int r = i >> 8, c = i & 255;
        xbf[r * XBF_STRIDE + c] = (__bf16)xs[r * 257 + c];
    }
    __syncthreads();   // xbf ready; xs dead (reused as hs below)

    const int wave = tid >> 5;
    const int lane = tid & 31;
    const int lo   = lane & 15;   // A row / B column within tile
    const int hi   = lane >> 4;   // lane half selects K sub-range

    const __bf16* xrL = &xbf[lo * XBF_STRIDE];          // rows 0..15
    const __bf16* xrH = &xbf[(16 + lo) * XBF_STRIDE];   // rows 16..31
    const __bf16* kTl0 = kT + (size_t)((wave * 2 + 0) * 16 + lo) * DD;
    const __bf16* kTl1 = kT + (size_t)((wave * 2 + 1) * 16 + lo) * DD;

    v8f acc00 = {}; v8f acc01 = {};   // rows 0-15  x {nt0, nt1}
    v8f acc10 = {}; v8f acc11 = {};   // rows 16-31 x {nt0, nt1}
    #pragma unroll
    for (int kt = 0; kt < 8; ++kt) {
        const int kb = kt * 32;
        // A fragments: 16x32 bf16; lane half 'hi' holds K runs
        // [kb+hi*8, +8) and [kb+16+hi*8, +8) -> two aligned b128 LDS loads.
        v8bf aL0 = *(const v8bf*)(xrL + kb + hi * 8);
        v8bf aL1 = *(const v8bf*)(xrL + kb + 16 + hi * 8);
        v8bf aH0 = *(const v8bf*)(xrH + kb + hi * 8);
        v8bf aH1 = *(const v8bf*)(xrH + kb + 16 + hi * 8);
        v16bf aLo = __builtin_shufflevector(aL0, aL1,
                        0,1,2,3,4,5,6,7,8,9,10,11,12,13,14,15);
        v16bf aHi = __builtin_shufflevector(aH0, aH1,
                        0,1,2,3,4,5,6,7,8,9,10,11,12,13,14,15);
        // B fragments: 32x16 bf16, lane half hi holds K runs
        // [kb+hi*8, +8) and [kb+16+hi*8, +8) in kT[n][k] (same as A).
        v16bf b0, b1;
        #pragma unroll
        for (int e = 0; e < 16; ++e) b0[e] = kTl0[kb + hi * 8 + (e & 7) + ((e >> 3) << 4)];
        #pragma unroll
        for (int e = 0; e < 16; ++e) b1[e] = kTl1[kb + hi * 8 + (e & 7) + ((e >> 3) << 4)];

        acc00 = __builtin_amdgcn_wmma_f32_16x16x32_bf16(
                    false, aLo, false, b0, (short)0, acc00, false, false);
        acc01 = __builtin_amdgcn_wmma_f32_16x16x32_bf16(
                    false, aLo, false, b1, (short)0, acc01, false, false);
        acc10 = __builtin_amdgcn_wmma_f32_16x16x32_bf16(
                    false, aHi, false, b0, (short)0, acc10, false, false);
        acc11 = __builtin_amdgcn_wmma_f32_16x16x32_bf16(
                    false, aHi, false, b1, (short)0, acc11, false, false);
    }

    // C/D layout: VGPR r -> row (r + 8*hi), col = lo within N-tile.
    const int c0 = (wave * 2 + 0) * 16 + lo;
    const int c1 = (wave * 2 + 1) * 16 + lo;
    #pragma unroll
    for (int r = 0; r < 8; ++r) {
        int row = r + 8 * hi;
        hs[row * 257 + c0]        = fast_tanh(acc00[r]);
        hs[row * 257 + c1]        = fast_tanh(acc01[r]);
        hs[(16 + row) * 257 + c0] = fast_tanh(acc10[r]);
        hs[(16 + row) * 257 + c1] = fast_tanh(acc11[r]);
    }
    __syncthreads();

    // Mini-GEMM: (32 x 256) @ (256 x 8) -> 256 outputs, one per thread.
    // Output layout transposed to [b][h][s] so softmax/pool are coalesced.
    {
        int mr = tid >> 3, head = tid & 7;
        float s = 0.f;
        #pragma unroll 8
        for (int k = 0; k < DD; ++k)
            s = fmaf(hs[mr * 257 + k], os[k * HH + head], s);
        wraw[(size_t)(b * HH + head) * SS + srow + mr] = s;
    }
}

// ---------------------------------------------------------------------------
// Kernel 2: masked softmax over the sequence axis, one WG per (b, head).
// Scores are [b][h][s] -> fully coalesced. In-place (register-cached reads).
// ---------------------------------------------------------------------------
__global__ __launch_bounds__(256) void softmax_kernel(
    float* __restrict__ w, const unsigned char* __restrict__ mask) {

    __shared__ float red[256];
    const int bh  = blockIdx.x;           // b*8 + h
    const int b   = bh >> 3;
    const int tid = threadIdx.x;
    float* wrow = w + (size_t)bh * SS;
    const unsigned char* mrow = mask + (size_t)b * SS;

    float v[32];
    #pragma unroll
    for (int i = 0; i < 32; ++i) {
        int s = tid + i * 256;
        float val = wrow[s];
        if (!mrow[s]) val -= NEG_BIG;
        v[i] = val;
    }

    float mx = -INFINITY;
    #pragma unroll
    for (int i = 0; i < 32; ++i) mx = fmaxf(mx, v[i]);
    red[tid] = mx;
    __syncthreads();
    for (int off = 128; off > 0; off >>= 1) {
        if (tid < off) red[tid] = fmaxf(red[tid], red[tid + off]);
        __syncthreads();
    }
    mx = red[0];
    __syncthreads();

    float sum = 0.f;
    #pragma unroll
    for (int i = 0; i < 32; ++i) { v[i] = __expf(v[i] - mx); sum += v[i]; }
    red[tid] = sum;
    __syncthreads();
    for (int off = 128; off > 0; off >>= 1) {
        if (tid < off) red[tid] += red[tid + off];
        __syncthreads();
    }
    const float inv = __builtin_amdgcn_rcpf(red[0]);

    #pragma unroll
    for (int i = 0; i < 32; ++i) wrow[tid + i * 256] = v[i] * inv;
}

// ---------------------------------------------------------------------------
// Kernel 3: partial pooling. One WG per (batch, 512-seq chunk); thread owns
// one feature column d and accumulates all 8 head sums. Probs are [b][h][s].
// ---------------------------------------------------------------------------
__global__ __launch_bounds__(256) void pool_partial(
    const float* __restrict__ x,
    const float* __restrict__ p,
    float*       __restrict__ part) {

    __shared__ float ps[HH * 512];     // probs chunk [h][s], 16 KB
    const int blk = blockIdx.x;
    const int b   = blk >> 4;
    const int ck  = blk & 15;
    const int s0  = ck * 512;
    const int tid = threadIdx.x;       // feature column d

    for (int i = tid; i < HH * 512; i += 256) {
        int h = i >> 9, s = i & 511;
        ps[i] = p[(size_t)(b * HH + h) * SS + s0 + s];   // coalesced per h-row
    }
    __syncthreads();

    float acc[HH] = {};
    const float* xb = x + (size_t)(b * SS + s0) * DD + tid;
    for (int s = 0; s < 512; ++s) {
        float xv = xb[(size_t)s * DD];
        #pragma unroll
        for (int h = 0; h < HH; ++h)
            acc[h] = fmaf(ps[h * 512 + s], xv, acc[h]);  // LDS broadcast reads
    }
    #pragma unroll
    for (int h = 0; h < HH; ++h)
        part[(size_t)((b * 16 + ck) * HH + h) * DD + tid] = acc[h];
}

// ---------------------------------------------------------------------------
// Kernel 4: reduce 16 chunk-partials -> out[b][h*256 + d]
// ---------------------------------------------------------------------------
__global__ __launch_bounds__(256) void pool_reduce(
    const float* __restrict__ part, float* __restrict__ out) {
    const int idx = blockIdx.x * 256 + threadIdx.x;  // 16*8*256 = 32768
    const int b   = idx >> 11;
    const int rem = idx & 2047;                      // h*256 + d
    float s = 0.f;
    #pragma unroll
    for (int ck = 0; ck < 16; ++ck)
        s += part[(size_t)(b * 16 + ck) * 2048 + rem];
    out[idx] = s;
}

// ---------------------------------------------------------------------------
// Launch
// ---------------------------------------------------------------------------
extern "C" void kernel_launch(void* const* d_in, const int* in_sizes, int n_in,
                              void* d_out, int out_size, void* d_ws, size_t ws_size,
                              hipStream_t stream) {
    const float*         x    = (const float*)d_in[0];
    const unsigned char* mask = (const unsigned char*)d_in[1];  // jnp.bool_ = 1 byte
    const float*         kk   = (const float*)d_in[2];
    const float*         ok   = (const float*)d_in[3];
    float*               out  = (float*)d_out;

    // Workspace layout (bytes):
    //   [0, 128K)            : kT bf16 (256x256)
    //   [128K, 128K+4M)      : scores -> probs (B*H*S f32)
    //   [128K+4M, +2M)       : pooling partials (B*16*H*D f32)
    char*   ws     = (char*)d_ws;
    __bf16* kT     = (__bf16*)ws;
    float*  scores = (float*)(ws + (size_t)(1 << 17));
    float*  part   = (float*)(ws + (size_t)(1 << 17) + (size_t)BB * SS * HH * 4);

    cvt_kT        <<<DD,                 256, 0, stream>>>(kk, kT);
    score_kernel  <<<BB * SS / MROWS,    256, 0, stream>>>(x, kT, ok, scores);
    softmax_kernel<<<BB * HH,            256, 0, stream>>>(scores, mask);
    pool_partial  <<<BB * 16,            256, 0, stream>>>(x, scores, part);
    pool_reduce   <<<BB * HH * DD / 256, 256, 0, stream>>>(part, out);
}
